// MultiHeadSelfAttention_17523466567782
// MI455X (gfx1250) — matrix operand order repaired and checked by reading, not compiled, not run on hardware
//
#include <hip/hip_runtime.h>
#include <hip/hip_bf16.h>

// ---------------- problem constants (from reference setup) ----------------
constexpr int Bc = 2;
constexpr int Sc = 2048;
constexpr int Dc = 1024;
constexpr int Hc = 16;
constexpr int DK = 64;          // dk = D/H

// ---------------- WMMA types ----------------
typedef __attribute__((ext_vector_type(16))) __bf16 v16bf;
typedef __attribute__((ext_vector_type(8)))  float  v8f;
typedef __attribute__((ext_vector_type(4)))  unsigned int v4u;
typedef __attribute__((ext_vector_type(8)))  int v8i_t;
typedef __attribute__((ext_vector_type(4)))  int v4i_t;

union BF16Frag { v16bf v; unsigned int u[8]; };

// native f32 -> bf16
__device__ __forceinline__ unsigned short f2bf(float f) {
    return __builtin_bit_cast(unsigned short, (__bf16)f);
}

// A-matrix 16x32 bf16: lane m = lane%16, half = lane/16,
// vgpr p holds k = 2*p + 16*half, +{0,1}
__device__ __forceinline__ int kA(int p, int hf) {
    return 2 * p + 16 * hf;
}
// B-matrix 32x16 bf16: lane n = lane%16, half = lane/16,
// vgpr p holds k = 2*p + 16*half, +{0,1}
__device__ __forceinline__ int kB(int p, int hf) {
    return 2 * p + 16 * hf;
}

__device__ __forceinline__ v16bf load_frag_A(const unsigned short* __restrict__ base,
                                             int ld, int k0) {
    const int lane = threadIdx.x & 31;
    const int m = lane & 15, hf = lane >> 4;
    const unsigned short* row = base + (size_t)m * ld + k0;
    BF16Frag f;
#pragma unroll
    for (int p = 0; p < 8; ++p)
        f.u[p] = *(const unsigned int*)(row + kA(p, hf));
    return f.v;
}

__device__ __forceinline__ v16bf load_frag_B(const unsigned short* __restrict__ base,
                                             int ld, int k0) {
    const int lane = threadIdx.x & 31;
    const int n = lane & 15, hf = lane >> 4;
    const unsigned short* row = base + (size_t)n * ld + k0;
    BF16Frag f;
#pragma unroll
    for (int p = 0; p < 8; ++p)
        f.u[p] = *(const unsigned int*)(row + kB(p, hf));
    return f.v;
}

__device__ __forceinline__ v8f wmma_bf16(v16bf a, v16bf b, v8f c) {
    return __builtin_amdgcn_wmma_f32_16x16x32_bf16(false, a, false, b,
                                                   (short)0, c, false, false);
}

// CDNA5 async global->LDS copy (16B/lane), tracked by ASYNCcnt (fallback path).
__device__ __forceinline__ void async_ld16(void* lds_ptr, const void* gptr) {
    unsigned ldsoff = (unsigned)(unsigned long long)lds_ptr;
    asm volatile("global_load_async_to_lds_b128 %0, %1, off"
                 :: "v"(ldsoff), "v"(gptr) : "memory");
}

// ---------------- Tensor Data Mover (TDM) staging ---------------------------
#if defined(__has_builtin)
#  if __has_builtin(__builtin_amdgcn_tensor_load_to_lds) && \
      __has_builtin(__builtin_amdgcn_s_wait_tensorcnt)
#    define HAVE_TDM 1
#  endif
#endif
#ifndef HAVE_TDM
#  define HAVE_TDM 0
#endif

#if HAVE_TDM
// 2-D tile load: elements are 2 bytes (data_size code 1).
// D# packing per CDNA5 ISA ch.8: group0 = {count/type/addr}, group1 = dims.
// This toolchain's builtin takes 6 args: (g0 u32x4, g1 i32x8, i32x4, i32x4,
// i32x8, i32 cpol); the trailing groups are zero for 2-D tiles.
__device__ __forceinline__ void tdm_load_2d(void* lds_ptr, const void* gptr,
                                            unsigned tensor_d0, unsigned tensor_d1,
                                            unsigned tile_d0, unsigned tile_d1,
                                            unsigned stride0) {
    const unsigned lds = (unsigned)(unsigned long long)lds_ptr;
    const unsigned long long ga = (unsigned long long)gptr;
    v4u g0;
    g0[0] = 1u;                                         // count=1
    g0[1] = lds;                                        // lds_addr
    g0[2] = (unsigned)(ga & 0xFFFFFFFFu);               // global_addr[31:0]
    g0[3] = (unsigned)((ga >> 32) & 0x1FFFFFFu) | (2u << 30);  // addr[56:32]|type=2
    v8i_t g1;
    g1[0] = (int)(1u << 16);                            // data_size=1 (2 bytes)
    g1[1] = (int)((tensor_d0 & 0xFFFFu) << 16);         // tensor_dim0[15:0]
    g1[2] = (int)(((tensor_d0 >> 16) & 0xFFFFu) | ((tensor_d1 & 0xFFFFu) << 16));
    g1[3] = (int)(((tensor_d1 >> 16) & 0xFFFFu) | ((tile_d0 & 0xFFFFu) << 16));
    g1[4] = (int)(tile_d1 & 0xFFFFu);                   // tile_dim1 (tile_dim2=0)
    g1[5] = (int)stride0;                               // tensor_dim0_stride[31:0]
    g1[6] = 0;
    g1[7] = 0;
    const v4i_t gz4 = (v4i_t){0, 0, 0, 0};
    const v8i_t gz8 = (v8i_t){0, 0, 0, 0, 0, 0, 0, 0};
    __builtin_amdgcn_tensor_load_to_lds(g0, g1, gz4, gz4, gz8, 0);
}
#endif

// ---------------- kernel 1a: f32 -> bf16 elementwise ----------------
__global__ void cvt_bf16_kernel(const float* __restrict__ in,
                                unsigned short* __restrict__ out, int n) {
    int i = blockIdx.x * blockDim.x + threadIdx.x;
    if (i < n) out[i] = f2bf(in[i]);
}

// ---------------- kernel 1b: transpose [Hh][Kk][Nn] -> [Hh][Nn][Kk] ----------
__global__ void transpose_bf16_kernel(const float* __restrict__ in,
                                      unsigned short* __restrict__ out,
                                      int Hh, int Kk, int Nn) {
    int i = blockIdx.x * blockDim.x + threadIdx.x;
    int total = Hh * Kk * Nn;
    if (i >= total) return;
    int n = i % Nn;
    int k = (i / Nn) % Kk;
    int h = i / (Nn * Kk);
    out[((size_t)h * Nn + n) * Kk + k] = f2bf(in[i]);
}

// ---------------- kernel 2: QKV projection, 2x-unrolled pipeline ------------
__global__ __launch_bounds__(256) void qkv_kernel(
        const unsigned short* __restrict__ xb,
        const unsigned short* __restrict__ wqt,
        const unsigned short* __restrict__ wkt,
        const unsigned short* __restrict__ wvt,
        unsigned short* __restrict__ Qp,
        unsigned short* __restrict__ Kp,
        unsigned short* __restrict__ Vtg) {
    const int wave = (blockIdx.x * blockDim.x + threadIdx.x) >> 5;
    const int lane = threadIdx.x & 31;
    const int ST2 = Sc / 32;                     // 64
    const int perMat = Bc * Hc * ST2;            // 2048
    const int mat = wave / perMat;               // 0=Q 1=K 2=V
    const int rem = wave % perMat;
    const int st  = rem % ST2;
    const int bh  = rem / ST2;
    const int b = bh / Hc, h = bh % Hc;

    const unsigned short* A0 = xb + ((size_t)b * Sc + st * 32) * Dc;
    const unsigned short* A1 = A0 + (size_t)16 * Dc;
    const unsigned short* Wt = (mat == 0 ? wqt : (mat == 1 ? wkt : wvt))
                               + (size_t)h * DK * Dc;

    const int nn = lane & 15, hf = lane >> 4;
    const unsigned short* pf0 = A0 + (size_t)nn * Dc;
    const unsigned short* pf1 = A1 + (size_t)nn * Dc;

    v8f acc[8];
#pragma unroll
    for (int i = 0; i < 8; ++i) acc[i] = (v8f){};

    v16bf a0s0 = load_frag_A(A0, Dc, 0);
    v16bf a1s0 = load_frag_A(A1, Dc, 0);
    v16bf b0[4];
#pragma unroll
    for (int j = 0; j < 4; ++j) b0[j] = load_frag_B(Wt + (size_t)(16 * j) * Dc, Dc, 0);
    v16bf a0s1, a1s1, b1[4];

    for (int k0 = 0; k0 < Dc; k0 += 64) {
        const int k1 = k0 + 32;
        __builtin_prefetch(pf0 + k0 + 128, 0, 1);
        __builtin_prefetch(pf1 + k0 + 128, 0, 1);
        a0s1 = load_frag_A(A0, Dc, k1);
        a1s1 = load_frag_A(A1, Dc, k1);
#pragma unroll
        for (int j = 0; j < 4; ++j)
            b1[j] = load_frag_B(Wt + (size_t)(16 * j) * Dc, Dc, k1);
#pragma unroll
        for (int j = 0; j < 4; ++j) {
            acc[j]     = wmma_bf16(a0s0, b0[j], acc[j]);
            acc[4 + j] = wmma_bf16(a1s0, b0[j], acc[4 + j]);
        }
        const int k2 = (k0 + 64) & (Dc - 1);
        a0s0 = load_frag_A(A0, Dc, k2);
        a1s0 = load_frag_A(A1, Dc, k2);
#pragma unroll
        for (int j = 0; j < 4; ++j)
            b0[j] = load_frag_B(Wt + (size_t)(16 * j) * Dc, Dc, k2);
#pragma unroll
        for (int j = 0; j < 4; ++j) {
            acc[j]     = wmma_bf16(a0s1, b1[j], acc[j]);
            acc[4 + j] = wmma_bf16(a1s1, b1[j], acc[4 + j]);
        }
    }

    if (mat < 2) {   // Q (scaled by dk^-0.5), K: row-major [bh][s][dk]
        const float scl = (mat == 0) ? 0.125f : 1.0f;
#pragma unroll
        for (int t = 0; t < 2; ++t) {
            unsigned short* base = (mat == 0 ? Qp : Kp)
                + ((size_t)bh * Sc + st * 32 + 16 * t + 8 * hf) * DK + nn;
#pragma unroll
            for (int j = 0; j < 4; ++j)
#pragma unroll
                for (int r = 0; r < 8; ++r)
                    base[(size_t)r * DK + 16 * j] = f2bf(acc[4 * t + j][r] * scl);
        }
    } else {         // V transposed: [bh][dk][S]
#pragma unroll
        for (int t = 0; t < 2; ++t)
#pragma unroll
            for (int j = 0; j < 4; ++j) {
                unsigned short* dst = Vtg + ((size_t)bh * DK + 16 * j + nn) * Sc
                                      + st * 32 + 16 * t + 8 * hf;
#pragma unroll
                for (int r = 0; r < 8; ++r) dst[r] = f2bf(acc[4 * t + j][r]);
            }
    }
}

// ---------------- kernel 3: flash attention -------------------------------
// 4 waves/WG, 64 q-rows per WG. K/V chunks staged into double-buffered LDS by
// the Tensor Data Mover (wave 0 issues 2 descriptors/chunk, TENSORcnt-tracked)
// or, if TDM builtins are absent, per-lane async global->LDS DMA.
__global__ __launch_bounds__(128) void attn_kernel(
        const unsigned short* __restrict__ Qp,
        const unsigned short* __restrict__ Kp,
        const unsigned short* __restrict__ Vtg,
        unsigned short* __restrict__ concat) {
    __shared__ __align__(16) unsigned short kst[2][64 * 64];   // [s][k]  2x8KB
    __shared__ __align__(16) unsigned short vst[2][64 * 64];   // [n][s]  2x8KB
    __shared__ __align__(16) unsigned short pst[4][16 * 64];   // per-wave P 8KB

    const int tid = threadIdx.x;
    const int wave = tid >> 5, lane = tid & 31;
    const int nn = lane & 15, hf = lane >> 4;
    const int QB = Sc / 64;                      // 32 q-blocks per (b,h)
    const int qb = blockIdx.x % QB;
    const int bh = blockIdx.x / QB;
    const int b = bh / Hc, h = bh % Hc;
    const int qt = qb * 64 + wave * 16;

    const unsigned short* Kg = Kp  + (size_t)bh * Sc * DK;   // [s][64]
    const unsigned short* Vg = Vtg + (size_t)bh * DK * Sc;   // [n][S]

    const unsigned short* Qbase = Qp + ((size_t)bh * Sc + qt) * DK;
    const v16bf qa0 = load_frag_A(Qbase, DK, 0);
    const v16bf qa1 = load_frag_A(Qbase, DK, 32);

    // all-ones B fragment: P x 1s -> every lane holds its row's sum
    BF16Frag of;
#pragma unroll
    for (int p = 0; p < 8; ++p) of.u[p] = 0x3F803F80u;
    const v16bf onesb = of.v;

    v8f o[4], o4;
#pragma unroll
    for (int j = 0; j < 4; ++j) o[j] = (v8f){};
    o4 = (v8f){};
    float mrow[8];
#pragma unroll
    for (int r = 0; r < 8; ++r) mrow[r] = -3.0e38f;

    unsigned short* pw = pst[wave];

    auto stage = [&](int buf, int s0) {
#if HAVE_TDM
        if (wave == 0) {
            // K chunk: 64x64 contiguous (1-D tile, 4096 elems); V: 2-D 64x64
            tdm_load_2d(&kst[buf][0], Kg + (size_t)s0 * DK,
                        4096u, 1u, 4096u, 0u, 4096u);
            tdm_load_2d(&vst[buf][0], Vg + s0,
                        (unsigned)Sc, 64u, 64u, 64u, (unsigned)Sc);
        }
#else
#pragma unroll
        for (int i = 0; i < 4; ++i) {
            const int idx = tid + 128 * i;
            async_ld16(&kst[buf][idx * 8], Kg + (size_t)s0 * DK + idx * 8);
        }
#pragma unroll
        for (int i = 0; i < 4; ++i) {
            const int idx = tid + 128 * i;
            const int row = idx >> 3, col = idx & 7;
            async_ld16(&vst[buf][idx * 8], Vg + (size_t)row * Sc + s0 + col * 8);
        }
#endif
    };

    const int NC = Sc / 64;                      // 32 chunks
    stage(0, 0);
    for (int ci = 0; ci < NC; ++ci) {
        const int p = ci & 1;
        if (ci) __syncthreads();                 // buf p^1 free to overwrite
        if (ci + 1 < NC) {
            stage(p ^ 1, (ci + 1) * 64);
#if HAVE_TDM
            __builtin_amdgcn_s_wait_tensorcnt(2);    // chunk ci's 2 loads landed
#else
            asm volatile("s_wait_asynccnt 0x8" ::: "memory");
#endif
        } else {
#if HAVE_TDM
            __builtin_amdgcn_s_wait_tensorcnt(0);
#else
            asm volatile("s_wait_asynccnt 0x0" ::: "memory");
#endif
        }
        __syncthreads();                         // all waves' data visible

        // ---- scores: 16(q) x 64(s), 8 WMMAs from LDS ----
        v8f c[4];
#pragma unroll
        for (int j = 0; j < 4; ++j) {
            v8f t = {};
            t = wmma_bf16(qa0, load_frag_B(&kst[p][(16 * j) * 64], 64, 0),  t);
            t = wmma_bf16(qa1, load_frag_B(&kst[p][(16 * j) * 64], 64, 32), t);
            c[j] = t;
        }

        // ---- online softmax: max tree + exp (sum comes from WMMA) ----
        float alpha[8];
#pragma unroll
        for (int r = 0; r < 8; ++r) {
            float t = fmaxf(fmaxf(c[0][r], c[1][r]), fmaxf(c[2][r], c[3][r]));
#pragma unroll
            for (int msk = 1; msk < 16; msk <<= 1)
                t = fmaxf(t, __shfl_xor(t, msk, 32));
            const float mn = fmaxf(mrow[r], t);
            alpha[r] = __expf(mrow[r] - mn);
            mrow[r] = mn;
#pragma unroll
            for (int j = 0; j < 4; ++j) c[j][r] = __expf(c[j][r] - mn);
        }

        // ---- stage P (wave-private LDS, in-order within wave) ----
#pragma unroll
        for (int r = 0; r < 8; ++r)
#pragma unroll
            for (int j = 0; j < 4; ++j)
                pw[(r + 8 * hf) * 64 + 16 * j + nn] = f2bf(c[j][r]);
        const v16bf pa0 = load_frag_A(pw, 64, 0);
        const v16bf pa1 = load_frag_A(pw, 64, 32);

        // ---- rescale O and row-sums, accumulate P x V and P x 1s ----
#pragma unroll
        for (int j = 0; j < 4; ++j)
#pragma unroll
            for (int r = 0; r < 8; ++r) o[j][r] *= alpha[r];
#pragma unroll
        for (int r = 0; r < 8; ++r) o4[r] *= alpha[r];
#pragma unroll
        for (int j = 0; j < 4; ++j) {
            o[j] = wmma_bf16(pa0, load_frag_B(&vst[p][(16 * j) * 64], 64, 0),  o[j]);
            o[j] = wmma_bf16(pa1, load_frag_B(&vst[p][(16 * j) * 64], 64, 32), o[j]);
        }
        o4 = wmma_bf16(pa0, onesb, o4);
        o4 = wmma_bf16(pa1, onesb, o4);
    }

    // ---- finalize: concat[b][s][h*dk + n] bf16 ----
    unsigned short* dst = concat + ((size_t)b * Sc + qt) * Dc + h * DK;
#pragma unroll
    for (int r = 0; r < 8; ++r) {
        const float inv = 1.0f / o4[r];
        const size_t row = (size_t)(r + 8 * hf) * Dc;
#pragma unroll
        for (int j = 0; j < 4; ++j)
            dst[row + 16 * j + nn] = f2bf(o[j][r] * inv);
    }
}

// ---------------- kernel 4: output projection, 2x-unrolled pipeline ---------
__global__ __launch_bounds__(256) void outproj_kernel(
        const unsigned short* __restrict__ concat,
        const unsigned short* __restrict__ wot,
        float* __restrict__ out) {
    const int wave = (blockIdx.x * blockDim.x + threadIdx.x) >> 5;
    const int lane = threadIdx.x & 31;
    const int NT = Dc / 64, ST2 = Sc / 32;       // 16, 64
    const int nt = wave % NT;
    const int st = (wave / NT) % ST2;
    const int b  = wave / (NT * ST2);

    const unsigned short* A0 = concat + ((size_t)b * Sc + st * 32) * Dc;
    const unsigned short* A1 = A0 + (size_t)16 * Dc;
    const unsigned short* Bt = wot + (size_t)nt * 64 * Dc;

    const int nn = lane & 15, hf = lane >> 4;
    const unsigned short* pf0 = A0 + (size_t)nn * Dc;
    const unsigned short* pf1 = A1 + (size_t)nn * Dc;

    v8f acc[8];
#pragma unroll
    for (int i = 0; i < 8; ++i) acc[i] = (v8f){};

    v16bf a0s0 = load_frag_A(A0, Dc, 0);
    v16bf a1s0 = load_frag_A(A1, Dc, 0);
    v16bf b0[4];
#pragma unroll
    for (int j = 0; j < 4; ++j) b0[j] = load_frag_B(Bt + (size_t)(16 * j) * Dc, Dc, 0);
    v16bf a0s1, a1s1, b1[4];

    for (int k0 = 0; k0 < Dc; k0 += 64) {
        const int k1 = k0 + 32;
        __builtin_prefetch(pf0 + k0 + 128, 0, 1);
        __builtin_prefetch(pf1 + k0 + 128, 0, 1);
        a0s1 = load_frag_A(A0, Dc, k1);
        a1s1 = load_frag_A(A1, Dc, k1);
#pragma unroll
        for (int j = 0; j < 4; ++j)
            b1[j] = load_frag_B(Bt + (size_t)(16 * j) * Dc, Dc, k1);
#pragma unroll
        for (int j = 0; j < 4; ++j) {
            acc[j]     = wmma_bf16(a0s0, b0[j], acc[j]);
            acc[4 + j] = wmma_bf16(a1s0, b0[j], acc[4 + j]);
        }
        const int k2 = (k0 + 64) & (Dc - 1);
        a0s0 = load_frag_A(A0, Dc, k2);
        a1s0 = load_frag_A(A1, Dc, k2);
#pragma unroll
        for (int j = 0; j < 4; ++j)
            b0[j] = load_frag_B(Bt + (size_t)(16 * j) * Dc, Dc, k2);
#pragma unroll
        for (int j = 0; j < 4; ++j) {
            acc[j]     = wmma_bf16(a0s1, b1[j], acc[j]);
            acc[4 + j] = wmma_bf16(a1s1, b1[j], acc[4 + j]);
        }
    }

#pragma unroll
    for (int t = 0; t < 2; ++t) {
        float* dst = out + ((size_t)b * Sc + st * 32 + 16 * t + 8 * hf) * Dc
                     + nt * 64 + nn;
#pragma unroll
        for (int j = 0; j < 4; ++j)
#pragma unroll
            for (int r = 0; r < 8; ++r)
                dst[(size_t)r * Dc + 16 * j] = acc[4 * t + j][r];
    }
}

// ---------------- host launch ----------------
extern "C" void kernel_launch(void* const* d_in, const int* in_sizes, int n_in,
                              void* d_out, int out_size, void* d_ws, size_t ws_size,
                              hipStream_t stream) {
    (void)in_sizes; (void)n_in; (void)out_size; (void)ws_size;
    const float* x  = (const float*)d_in[0];
    const float* WQ = (const float*)d_in[1];
    const float* WK = (const float*)d_in[2];
    const float* WV = (const float*)d_in[3];
    const float* WO = (const float*)d_in[4];
    float* out = (float*)d_out;

    char* w = (char*)d_ws;
    size_t off = 0;
    unsigned short* xb  = (unsigned short*)(w + off); off += (size_t)Bc * Sc * Dc * 2;
    unsigned short* wqt = (unsigned short*)(w + off); off += (size_t)Hc * DK * Dc * 2;
    unsigned short* wkt = (unsigned short*)(w + off); off += (size_t)Hc * DK * Dc * 2;
    unsigned short* wvt = (unsigned short*)(w + off); off += (size_t)Hc * DK * Dc * 2;
    unsigned short* wot = (unsigned short*)(w + off); off += (size_t)Dc * Dc * 2;
    unsigned short* Qp  = (unsigned short*)(w + off); off += (size_t)Bc * Hc * Sc * DK * 2;
    unsigned short* Kp  = (unsigned short*)(w + off); off += (size_t)Bc * Hc * Sc * DK * 2;
    unsigned short* Vt  = (unsigned short*)(w + off); off += (size_t)Bc * Hc * Sc * DK * 2;
    unsigned short* cc  = (unsigned short*)(w + off); off += (size_t)Bc * Sc * Dc * 2;

    {   // conversions / transposes
        int n = Bc * Sc * Dc;
        cvt_bf16_kernel<<<(n + 255) / 256, 256, 0, stream>>>(x, xb, n);
        int nw = Hc * Dc * DK;
        transpose_bf16_kernel<<<(nw + 255) / 256, 256, 0, stream>>>(WQ, wqt, Hc, Dc, DK);
        transpose_bf16_kernel<<<(nw + 255) / 256, 256, 0, stream>>>(WK, wkt, Hc, Dc, DK);
        transpose_bf16_kernel<<<(nw + 255) / 256, 256, 0, stream>>>(WV, wvt, Hc, Dc, DK);
        int no = Dc * Dc;
        transpose_bf16_kernel<<<(no + 255) / 256, 256, 0, stream>>>(WO, wot, 1, Dc, Dc);
    }
    {   // QKV: 3 * B*H * (S/32) = 6144 waves, 8 waves/block
        int waves = 3 * Bc * Hc * (Sc / 32);
        qkv_kernel<<<waves / 8, 256, 0, stream>>>(xb, wqt, wkt, wvt, Qp, Kp, Vt);
    }
    {   // attention: 4-wave blocks, 64 q-rows per block
        int blocks = Bc * Hc * (Sc / 64);        // 1024
        attn_kernel<<<blocks, 128, 0, stream>>>(Qp, Kp, Vt, cc);
    }
    {   // output projection: B * (S/32) * (D/64) = 2048 waves
        int waves = Bc * (Sc / 32) * (Dc / 64);
        outproj_kernel<<<waves / 8, 256, 0, stream>>>(cc, wot, out);
    }
}
